// GCNLSTM_41532333753035
// MI455X (gfx1250) — matrix operand order repaired and checked by reading, not compiled, not run on hardware
//
#include <hip/hip_runtime.h>
#include <math.h>

typedef __attribute__((ext_vector_type(16))) __bf16 v16bf;
typedef __attribute__((ext_vector_type(8)))  float  v8f;

#define NN      100000      // N_NODES
#define NE      1600000     // N_EDGES
#define DIN     64          // INPUT_DIM
#define DGCN    128         // GCN_DIM
#define NG      200         // N_GRAPHS
#define NPG     500         // NODES_PER_GRAPH
#define SEQL    20
#define HID     128         // LSTM_DIM
#define BATCH   (NG - SEQL + 1)   // 181

union FragBF {
  v16bf v;
  uint4 q[2];
  unsigned short s[16];
};

__device__ __forceinline__ unsigned short f2bf(float f) {
  unsigned int u = __float_as_uint(f);
  u += 0x7fffu + ((u >> 16) & 1u);          // round to nearest even
  return (unsigned short)(u >> 16);
}

__device__ __forceinline__ float sigm(float x) { return 1.0f / (1.0f + __expf(-x)); }

// ---------------- utility ----------------
__global__ void k_zero_f32(float* p, int n) {
  for (int i = blockIdx.x * blockDim.x + threadIdx.x; i < n; i += gridDim.x * blockDim.x)
    p[i] = 0.0f;
}

// ---------------- degrees ----------------
__global__ void k_degrees(const int* __restrict__ src, const int* __restrict__ dst,
                          float* __restrict__ outdeg, float* __restrict__ indeg) {
  int e = blockIdx.x * blockDim.x + threadIdx.x;
  if (e < NE) {
    atomicAdd(&outdeg[src[e]], 1.0f);
    atomicAdd(&indeg[dst[e]], 1.0f);
  }
}

// h_scaled = x * out_deg^-1/2
__global__ void k_scale_src(const float* __restrict__ x, const float* __restrict__ outdeg,
                            float* __restrict__ hsc) {
  int i = blockIdx.x * blockDim.x + threadIdx.x;
  if (i < NN * DIN) {
    int node = i >> 6;
    hsc[i] = x[i] * rsqrtf(fmaxf(outdeg[node], 1.0f));
  }
}

// agg[dst] += h_scaled[src]   (16 lanes per edge, float4 gather, 4 atomics)
__global__ void k_edge_agg(const int* __restrict__ src, const int* __restrict__ dst,
                           const float* __restrict__ hsc, float* __restrict__ agg) {
  int t = blockIdx.x * blockDim.x + threadIdx.x;
  int e = t >> 4;
  int q = t & 15;
  if (e < NE) {
    int s = src[e], d = dst[e];
    const float4 v = *(const float4*)(hsc + s * DIN + q * 4);
    float* ap = agg + d * DIN + q * 4;
    atomicAdd(ap + 0, v.x);
    atomicAdd(ap + 1, v.y);
    atomicAdd(ap + 2, v.z);
    atomicAdd(ap + 3, v.w);
  }
}

// h_bf = bf16( agg * in_deg^-1/2 )
__global__ void k_finalize_bf(const float* __restrict__ agg, const float* __restrict__ indeg,
                              unsigned short* __restrict__ hbf) {
  int i = blockIdx.x * blockDim.x + threadIdx.x;
  if (i < NN * DIN) {
    int node = i >> 6;
    hbf[i] = f2bf(agg[i] * rsqrtf(fmaxf(indeg[node], 1.0f)));
  }
}

// wT_bf[n][k] = bf16(w_gcn[k][n])    (64x128 -> 128x64, K contiguous per column)
__global__ void k_conv_wgcn(const float* __restrict__ wg, unsigned short* __restrict__ wt) {
  int i = blockIdx.x * blockDim.x + threadIdx.x;  // over 128*64
  if (i < DGCN * DIN) {
    int n = i / DIN, k = i % DIN;
    wt[i] = f2bf(wg[k * DGCN + n]);
  }
}

// w_ih (512x128, row-major, K contiguous per row) -> bf16 same layout
__global__ void k_conv_wih(const float* __restrict__ w, unsigned short* __restrict__ wb) {
  int i = blockIdx.x * blockDim.x + threadIdx.x;
  if (i < 4 * HID * DGCN) wb[i] = f2bf(w[i]);
}

// ---------------- GCN GEMM (WMMA bf16) ----------------
// h_relu[NN,128] = relu(h_bf[NN,64] @ wT_bf^T + b_gcn)
// grid.x = 6250 M-tiles, 8 waves/block -> one 16x16 N-tile each
__global__ void __launch_bounds__(256) k_gcn_gemm(const unsigned short* __restrict__ hbf,
                                                  const unsigned short* __restrict__ wt,
                                                  const float* __restrict__ bias,
                                                  float* __restrict__ out) {
  int mtile = blockIdx.x;
  int wave = threadIdx.x >> 5;
  int lane = threadIdx.x & 31;
  int half = lane >> 4;
  int l16 = lane & 15;
  int row = mtile * 16 + l16;
  int ncol = wave * 16 + l16;

  v8f acc = {};
#pragma unroll
  for (int kc = 0; kc < DIN; kc += 32) {
    FragBF a, b;
    // A 16x32 bf16: lanes0-15 K {0..7,16..23}, lanes16-31 K {8..15,24..31}
    int ka = kc + half * 8;
    a.q[0] = *(const uint4*)(hbf + row * DIN + ka);
    a.q[1] = *(const uint4*)(hbf + row * DIN + ka + 16);
    // B 32x16 bf16: lanes0-15 K {0..7,16..23}, lanes16-31 K {8..15,24..31}
    int kb = kc + half * 8;
    b.q[0] = *(const uint4*)(wt + ncol * DIN + kb);
    b.q[1] = *(const uint4*)(wt + ncol * DIN + kb + 16);
    acc = __builtin_amdgcn_wmma_f32_16x16x32_bf16(false, a.v, false, b.v,
                                                  (short)0, acc, false, false);
  }
  float bn = bias[ncol];
#pragma unroll
  for (int r = 0; r < 8; ++r) {
    int m = mtile * 16 + r + half * 8;
    float v = acc[r] + bn;
    out[m * DGCN + ncol] = v > 0.0f ? v : 0.0f;
  }
}

// ---------------- per-graph mean pool -> bf16 ----------------
__global__ void k_pool(const float* __restrict__ hrelu, unsigned short* __restrict__ hgbf) {
  int g = blockIdx.x;        // 200
  int n = threadIdx.x;       // 128
  const float* p = hrelu + (size_t)g * NPG * DGCN + n;
  float s = 0.0f;
  for (int i = 0; i < NPG; ++i) s += p[i * DGCN];
  hgbf[g * DGCN + n] = f2bf(s * (1.0f / (float)NPG));
}

// ---------------- input projection GEMM (WMMA bf16) ----------------
// xp[200,512] = hg_bf[200,128] @ w_ih^T + b_ih    (xp[t] shared by all windows)
// grid (13, 4), 8 waves/block; M padded to 208 with guards
__global__ void __launch_bounds__(256) k_xp_gemm(const unsigned short* __restrict__ hgbf,
                                                 const unsigned short* __restrict__ wih,
                                                 const float* __restrict__ bih,
                                                 float* __restrict__ xp) {
  int mtile = blockIdx.x;                           // 0..12
  int ntile = blockIdx.y * 8 + (threadIdx.x >> 5);  // 0..31
  int lane = threadIdx.x & 31;
  int half = lane >> 4;
  int l16 = lane & 15;
  int row = mtile * 16 + l16;
  int col = ntile * 16 + l16;
  bool rowok = row < NG;

  v8f acc = {};
#pragma unroll
  for (int kc = 0; kc < DGCN; kc += 32) {
    FragBF a, b;
    int ka = kc + half * 8;
    if (rowok) {
      a.q[0] = *(const uint4*)(hgbf + row * DGCN + ka);
      a.q[1] = *(const uint4*)(hgbf + row * DGCN + ka + 16);
    } else {
      a.q[0] = make_uint4(0, 0, 0, 0);
      a.q[1] = make_uint4(0, 0, 0, 0);
    }
    int kb = kc + half * 8;
    b.q[0] = *(const uint4*)(wih + col * DGCN + kb);
    b.q[1] = *(const uint4*)(wih + col * DGCN + kb + 16);
    acc = __builtin_amdgcn_wmma_f32_16x16x32_bf16(false, a.v, false, b.v,
                                                  (short)0, acc, false, false);
  }
  float bn = bih[col];
#pragma unroll
  for (int r = 0; r < 8; ++r) {
    int m = mtile * 16 + r + half * 8;
    if (m < NG) xp[m * (4 * HID) + col] = acc[r] + bn;
  }
}

// ---------------- LSTM + FC ----------------
// one block per sequence window (181), h/c in LDS, 20 serial steps
__global__ void __launch_bounds__(128) k_lstm(const float* __restrict__ xp,
                                              const float* __restrict__ whh,
                                              const float* __restrict__ bhh,
                                              const float* __restrict__ wfc,
                                              const float* __restrict__ bfc,
                                              float* __restrict__ out) {
  __shared__ float hsh[HID];
  __shared__ float csh[HID];
  __shared__ float red[HID];
  int b = blockIdx.x;
  int n = threadIdx.x;
  hsh[n] = 0.0f;
  csh[n] = 0.0f;
  __syncthreads();

  for (int l = 0; l < SEQL; ++l) {
    int t = b + l;
    float g[4];
#pragma unroll
    for (int q = 0; q < 4; ++q) {
      int rowg = q * HID + n;
      float acc = xp[t * (4 * HID) + rowg] + bhh[rowg];
      const float* w = whh + rowg * HID;
      for (int k = 0; k < HID; ++k) acc = fmaf(w[k], hsh[k], acc);
      g[q] = acc;
    }
    __syncthreads();  // everyone done reading hsh
    float ig = sigm(g[0]);
    float fg = sigm(g[1]);
    float gg = tanhf(g[2]);
    float og = sigm(g[3]);
    float c = fg * csh[n] + ig * gg;
    csh[n] = c;
    hsh[n] = og * tanhf(c);
    __syncthreads();
  }

  red[n] = hsh[n] * wfc[n];
  __syncthreads();
  for (int s = 64; s > 0; s >>= 1) {
    if (n < s) red[n] += red[n + s];
    __syncthreads();
  }
  if (n == 0) out[b] = red[0] + bfc[0];
}

extern "C" void kernel_launch(void* const* d_in, const int* in_sizes, int n_in,
                              void* d_out, int out_size, void* d_ws, size_t ws_size,
                              hipStream_t stream) {
  (void)in_sizes; (void)n_in; (void)out_size; (void)ws_size;
  const float* x     = (const float*)d_in[0];
  const int*   src   = (const int*)d_in[1];
  const int*   dst   = (const int*)d_in[2];
  /* d_in[3] graph_ids: implicit via node index / NPG */
  const float* w_gcn = (const float*)d_in[4];
  const float* b_gcn = (const float*)d_in[5];
  const float* w_ih  = (const float*)d_in[6];
  const float* w_hh  = (const float*)d_in[7];
  const float* b_ih  = (const float*)d_in[8];
  const float* b_hh  = (const float*)d_in[9];
  const float* w_fc  = (const float*)d_in[10];
  const float* b_fc  = (const float*)d_in[11];
  float* pred = (float*)d_out;

  char* ws = (char*)d_ws;
  // layout (bytes); h_relu reuses the h_scaled+agg region after they're dead
  float* h_scaled        = (float*)(ws + 0);              // NN*64*4  = 25,600,000
  float* agg             = (float*)(ws + 25600000);       // NN*64*4
  float* h_relu          = (float*)(ws + 0);              // NN*128*4 = 51,200,000
  unsigned short* h_bf   = (unsigned short*)(ws + 51200000); // NN*64*2
  float* out_deg         = (float*)(ws + 64000000);       // NN*4
  float* in_deg          = (float*)(ws + 64400000);       // NN*4
  unsigned short* wT_bf  = (unsigned short*)(ws + 64800000); // 128*64*2
  unsigned short* wih_bf = (unsigned short*)(ws + 64816384); // 512*128*2
  unsigned short* hg_bf  = (unsigned short*)(ws + 64947456); // 200*128*2
  float* xp              = (float*)(ws + 64998656);       // 200*512*4

  // accumulators must be re-zeroed every call (deterministic)
  k_zero_f32<<<4096, 256, 0, stream>>>(agg, NN * DIN);
  k_zero_f32<<<256, 256, 0, stream>>>(out_deg, 2 * NN);  // out_deg + in_deg contiguous

  k_degrees<<<(NE + 255) / 256, 256, 0, stream>>>(src, dst, out_deg, in_deg);
  k_scale_src<<<(NN * DIN + 255) / 256, 256, 0, stream>>>(x, out_deg, h_scaled);
  k_edge_agg<<<(NE * 16 + 255) / 256, 256, 0, stream>>>(src, dst, h_scaled, agg);
  k_finalize_bf<<<(NN * DIN + 255) / 256, 256, 0, stream>>>(agg, in_deg, h_bf);

  k_conv_wgcn<<<(DGCN * DIN + 255) / 256, 256, 0, stream>>>(w_gcn, wT_bf);
  k_conv_wih<<<(4 * HID * DGCN + 255) / 256, 256, 0, stream>>>(w_ih, wih_bf);

  k_gcn_gemm<<<NN / 16, 256, 0, stream>>>(h_bf, wT_bf, b_gcn, h_relu);
  k_pool<<<NG, 128, 0, stream>>>(h_relu, hg_bf);
  k_xp_gemm<<<dim3(13, 4), 256, 0, stream>>>(hg_bf, wih_bf, b_ih, xp);
  k_lstm<<<BATCH, 128, 0, stream>>>(xp, w_hh, b_hh, w_fc, b_fc, pred);
}
